// EnhancedNavigationLTC_85005992722782
// MI455X (gfx1250) — matrix operand order repaired and checked by reading, not compiled, not run on hardware
//
#include <hip/hip_runtime.h>
#include <hip/hip_bf16.h>

// ---------------------------------------------------------------------------
// Problem constants (from reference)
// ---------------------------------------------------------------------------
#define BATCH 512
#define TSTEPS 512
#define IN 128
#define UNITS 64
#define OUT 15
#define UNFOLDS 4
#define LTC_EPS 1e-8f

// Scan-kernel decomposition: 32 lanes = 8 batch x 4 reduction-quarters.
#define BT 8            // batch tile per block -> BATCH/BT = 64 blocks
#define XS_STRIDE 156   // padded x-row stride (floats): f(i) = i + 8*(i>>5)
#define VS_STRIDE 9     // padded v-row stride: banks = (16q + 9jj + b) % 64

typedef _Float16 v16h __attribute__((ext_vector_type(16)));
typedef float    v8f  __attribute__((ext_vector_type(8)));

// ---------------------------------------------------------------------------
// gfx1250 async global->LDS path (guarded; falls back to regular staging)
// Builtin's first param (from clang diagnostic) is:
//   '__attribute__((vector_size(4*sizeof(int)))) int __device__ *'
// i.e. an address_space(1) pointer to a 16-byte int vector; LDS side is AS3.
// ---------------------------------------------------------------------------
#if defined(__has_builtin)
#if __has_builtin(__builtin_amdgcn_global_load_async_to_lds_b128)
#define LTC_ASYNC_LDS 1
#endif
#endif

#if defined(LTC_ASYNC_LDS)
typedef int ltc_v4i __attribute__((vector_size(16)));
typedef __attribute__((address_space(1))) ltc_v4i* ltc_gptr16;
typedef __attribute__((address_space(3))) ltc_v4i* ltc_lptr16;

__device__ __forceinline__ void ltc_async_copy16(float* lds_dst, const float* gsrc) {
    __builtin_amdgcn_global_load_async_to_lds_b128(
        (ltc_gptr16)const_cast<float*>(gsrc),
        (ltc_lptr16)lds_dst,
        0, 0);
}
__device__ __forceinline__ void ltc_wait_async() {
    asm volatile("s_wait_asynccnt 0x0" ::: "memory");
}
#endif

// ---------------------------------------------------------------------------
// Fast device math (v_exp_f32 / v_rcp_f32 / v_rsq_f32)
// ---------------------------------------------------------------------------
__device__ __forceinline__ float fast_sigmoid(float z) {
    float e = __builtin_amdgcn_exp2f(-z * 1.44269504088896340736f);
    return __builtin_amdgcn_rcpf(1.0f + e);
}

__device__ __forceinline__ float softplus_f(float v) {
    if (v > 15.0f) return v;
    return __logf(1.0f + __expf(v));
}

// combine partial sums across the 4 quarter-wave reduction groups (bits 3,4)
__device__ __forceinline__ float qreduce(float v) {
    v += __shfl_xor(v, 8, 32);
    v += __shfl_xor(v, 16, 32);
    return v;
}

// ---------------------------------------------------------------------------
// Kernel 0: parameter preprocessing (softplus + fusion of affine input map)
//   sens4[i*U+u] = { ss*input_w[i],  ss*(input_b[i]-smu),  sp(sw)*serev,  sp(sw) }
//   rec4 [j*U+u] = { sig,           -sig*mu,               sp(w)*erev,    sp(w)  }
//   neur4[u]     = { sp(cm)*UNFOLDS, sp(gleak), sp(gleak)*vleak, 0 }
// ---------------------------------------------------------------------------
__global__ __launch_bounds__(256) void ltc_prep_kernel(
    const float* __restrict__ input_w, const float* __restrict__ input_b,
    const float* __restrict__ sw,  const float* __restrict__ smu,
    const float* __restrict__ ssg, const float* __restrict__ serev,
    const float* __restrict__ w,   const float* __restrict__ mu,
    const float* __restrict__ sg,  const float* __restrict__ erev,
    const float* __restrict__ gleak, const float* __restrict__ vleak,
    const float* __restrict__ cm,
    float4* __restrict__ sens4, float4* __restrict__ rec4,
    float4* __restrict__ neur4)
{
    int idx = blockIdx.x * blockDim.x + threadIdx.x;
    const int NS = IN * UNITS;       // 8192
    const int NR = UNITS * UNITS;    // 4096
    if (idx < NS) {
        int i = idx >> 6;
        float A   = softplus_f(sw[idx]);
        float sig = ssg[idx];
        float c1  = sig * input_w[i];
        float c0  = sig * (input_b[i] - smu[idx]);
        sens4[idx] = make_float4(c1, c0, A * serev[idx], A);
    } else if (idx < NS + NR) {
        int r = idx - NS;
        float A   = softplus_f(w[r]);
        float sig = sg[r];
        rec4[r] = make_float4(sig, -sig * mu[r], A * erev[r], A);
    } else if (idx < NS + NR + UNITS) {
        int u = idx - NS - NR;
        float gp = softplus_f(gleak[u]);
        neur4[u] = make_float4(softplus_f(cm[u]) * (float)UNFOLDS,
                               gp, gp * vleak[u], 0.0f);
    }
}

// ---------------------------------------------------------------------------
// Kernel 1: sequential LTC scan.
//   grid = BATCH/8 = 64 blocks; block = 1024 threads = 32 waves (wave32).
//   lane = (b in 0..7, q in 0..3): b = batch element, q = reduction quarter.
//   wave owns units {2*wave, 2*wave+1}.
//   Per timestep: sensory (each lane: 2u x 32i) + 4 unfolds (2u x 16j),
//   partials combined with shfl_xor(8)+shfl_xor(16).
//   v state double-buffered in LDS -> 1 barrier per unfold.
//   x tile double-buffered in LDS via async global->LDS when available.
// ---------------------------------------------------------------------------
__global__ __launch_bounds__(1024) void ltc_scan_kernel(
    const float*  __restrict__ x,          // (B, T, I)
    const float4* __restrict__ sens4,      // (I*U)
    const float4* __restrict__ rec4,       // (U*U)
    const float4* __restrict__ neur4,      // (U)
    const float*  __restrict__ out_w, const float* __restrict__ out_b,
    const float*  __restrict__ ln_w,  const float* __restrict__ ln_b,
    _Float16*     __restrict__ hln)        // (B, U) f16, post-LayerNorm
{
    __shared__ __align__(16) float xs[2][BT * XS_STRIDE];  // x tiles (double buf)
    __shared__ float vsh[2][UNITS * VS_STRIDE];            // v state (double buf)

    const int tid  = threadIdx.x;
    const int lane = tid & 31;
    const int b    = lane & 7;                               // batch in tile
    const int q    = lane >> 3;                              // reduction quarter
    const int wave = __builtin_amdgcn_readfirstlane(tid >> 5);
    const int u0   = wave * 2;
    const int u1   = u0 + 1;
    const int b0   = blockIdx.x * BT;

    // init v = 0 (buffer 0 is the "start of timestep" buffer)
    for (int k = tid; k < UNITS * VS_STRIDE; k += 1024) {
        vsh[0][k] = 0.0f;
        vsh[1][k] = 0.0f;
    }

    // per-neuron constants (wave-uniform -> SGPRs)
    const float4 nc0 = neur4[u0];
    const float4 nc1 = neur4[u1];
    const float cmt0 = nc0.x, gp0 = nc0.y, gv0 = nc0.z;
    const float cmt1 = nc1.x, gp1 = nc1.y, gv1 = nc1.z;

    // waves 0..7 stage batch row `wave`; LDS offset is 16B aligned.
    const float* xrow = x + (size_t)(b0 + (wave & (BT - 1))) * (TSTEPS * IN);
    const int stg_off = wave * XS_STRIDE + lane * 4 + 8 * (lane >> 3);

    // prologue: stage t = 0 into buffer 0
    if (wave < BT) {
#if defined(LTC_ASYNC_LDS)
        ltc_async_copy16(&xs[0][stg_off], xrow + lane * 4);
        ltc_wait_async();
#else
        *(float4*)&xs[0][stg_off] = *(const float4*)(xrow + lane * 4);
#endif
    }
    __syncthreads();

    for (int t = 0; t < TSTEPS; ++t) {
        const float* xc = xs[t & 1];
        const bool has_next = (t + 1 < TSTEPS);

        // ---- kick staging of the next x tile into the other buffer ----
#if defined(LTC_ASYNC_LDS)
        if (has_next && wave < BT) {
            ltc_async_copy16(&xs[(t + 1) & 1][stg_off],
                             xrow + (t + 1) * IN + lane * 4);
            if (t + 2 < TSTEPS)
                __builtin_prefetch(xrow + (t + 2) * IN + lane * 4, 0, 3);
        }
#else
        float4 nx = make_float4(0.f, 0.f, 0.f, 0.f);
        if (has_next && wave < BT) {
            nx = *(const float4*)(xrow + (t + 1) * IN + lane * 4);
            if (t + 2 < TSTEPS)
                __builtin_prefetch(xrow + (t + 2) * IN + lane * 4, 0, 3);
        }
#endif

        // ---- sensory synapses: lane covers i in [q*32, q*32+32) ----
        float ns0 = 0.0f, ds0 = 0.0f, ns1 = 0.0f, ds1 = 0.0f;
        {
            const float* xsr = &xc[b * XS_STRIDE + q * 40];  // f(q*32+ii)=40q+ii
            const int ibase = q * 32;
            #pragma unroll 4
            for (int ii = 0; ii < 32; ++ii) {
                float xv = xsr[ii];
                float4 p0 = sens4[(ibase + ii) * UNITS + u0];
                float4 p1 = sens4[(ibase + ii) * UNITS + u1];
                float s0 = fast_sigmoid(fmaf(p0.x, xv, p0.y));
                ns0 = fmaf(p0.z, s0, ns0);
                ds0 = fmaf(p0.w, s0, ds0);
                float s1 = fast_sigmoid(fmaf(p1.x, xv, p1.y));
                ns1 = fmaf(p1.z, s1, ns1);
                ds1 = fmaf(p1.w, s1, ds1);
            }
        }
        ns0 = qreduce(ns0); ds0 = qreduce(ds0);
        ns1 = qreduce(ns1); ds1 = qreduce(ds1);

        // ---- ODE unfolds: lane covers j in [q*16, q*16+16) ----
        for (int r = 0; r < UNFOLDS; ++r) {
            const float* vr = vsh[r & 1];        // read buffer
            float*       vw = vsh[(r + 1) & 1];  // write buffer
            float n0 = 0.0f, d0 = 0.0f, n1 = 0.0f, d1 = 0.0f;
            const int jbase = q * 16;
            #pragma unroll 4
            for (int jj = 0; jj < 16; ++jj) {
                int j = jbase + jj;
                float vj = vr[j * VS_STRIDE + b];        // conflict-free
                float4 q0 = rec4[j * UNITS + u0];
                float4 q1 = rec4[j * UNITS + u1];
                float s0 = fast_sigmoid(fmaf(q0.x, vj, q0.y));
                n0 = fmaf(q0.z, s0, n0);
                d0 = fmaf(q0.w, s0, d0);
                float s1 = fast_sigmoid(fmaf(q1.x, vj, q1.y));
                n1 = fmaf(q1.z, s1, n1);
                d1 = fmaf(q1.w, s1, d1);
            }
            n0 = qreduce(n0); d0 = qreduce(d0);
            n1 = qreduce(n1); d1 = qreduce(d1);

            float v0o = vr[u0 * VS_STRIDE + b];          // broadcast reads
            float v1o = vr[u1 * VS_STRIDE + b];
            float num0 = fmaf(cmt0, v0o, gv0 + n0 + ns0);
            float den0 = cmt0 + gp0 + d0 + ds0 + LTC_EPS;
            float vn0  = num0 * __builtin_amdgcn_rcpf(den0);
            float num1 = fmaf(cmt1, v1o, gv1 + n1 + ns1);
            float den1 = cmt1 + gp1 + d1 + ds1 + LTC_EPS;
            float vn1  = num1 * __builtin_amdgcn_rcpf(den1);

            if (q == 0) {                     // one writer group per (u, b)
                vw[u0 * VS_STRIDE + b] = vn0;
                vw[u1 * VS_STRIDE + b] = vn1;
            }
            __syncthreads();                  // publish new v (single barrier)
        }

        // ---- publish the next x tile ----
        if (has_next) {
#if defined(LTC_ASYNC_LDS)
            if (wave < BT) ltc_wait_async();
#else
            if (wave < BT) *(float4*)&xs[(t + 1) & 1][stg_off] = nx;
#endif
            __syncthreads();
        }
    }

    // ---- epilogue: output affine + LayerNorm (wave 0, lanes 0..7) ----
    // after 512*4 unfolds the live buffer is vsh[0] (UNFOLDS even)
    if (wave == 0 && lane < BT) {
        float m = 0.0f, sq = 0.0f;
        for (int j = 0; j < UNITS; ++j) {
            float hv = fmaf(vsh[0][j * VS_STRIDE + lane], out_w[j], out_b[j]);
            m += hv; sq = fmaf(hv, hv, sq);
        }
        m *= (1.0f / UNITS);
        float var  = sq * (1.0f / UNITS) - m * m;
        float rstd = __builtin_amdgcn_rsqf(var + 1e-5f);
        for (int j = 0; j < UNITS; ++j) {
            float hv = fmaf(vsh[0][j * VS_STRIDE + lane], out_w[j], out_b[j]);
            float hn = fmaf((hv - m) * rstd, ln_w[j], ln_b[j]);
            hln[(size_t)(b0 + lane) * UNITS + j] = (_Float16)hn;
        }
    }
}

// ---------------------------------------------------------------------------
// Kernel 2: FC head via WMMA.  out = hln @ fc_w^T + fc_b
//   One wave per 16x16 output tile; K=64 as two v_wmma_f32_16x16x32_f16.
//   Fragment packing per CDNA5 ISA 7.12.2.
// ---------------------------------------------------------------------------
__global__ __launch_bounds__(32) void ltc_head_wmma_kernel(
    const _Float16* __restrict__ hln,   // (B, U) f16
    const float*    __restrict__ fc_w,  // (O, U) f32
    const float*    __restrict__ fc_b,  // (O)
    float*          __restrict__ out)   // (B, O)
{
    const int lane = threadIdx.x;       // full wave, EXEC all ones
    const int M    = lane & 15;
    const int hi   = lane >> 4;
    const int m0   = blockIdx.x * 16;

    v16h a0, a1, b0f, b1f;
    const int Nc = (M < OUT) ? M : (OUT - 1);   // clamp to stay in-bounds
    #pragma unroll
    for (int h = 0; h < 16; ++h) {
        int v  = h >> 1;
        int ka = ((v >= 4) ? 16 : 0) + (hi ? 8 : 0) + ((v & 3) << 1) + (h & 1);
        a0[h] = hln[(size_t)(m0 + M) * UNITS + ka];
        a1[h] = hln[(size_t)(m0 + M) * UNITS + 32 + ka];
        int kb = ka;
        float w0 = fc_w[Nc * UNITS + kb];
        float w1 = fc_w[Nc * UNITS + 32 + kb];
        b0f[h] = (M < OUT) ? (_Float16)w0 : (_Float16)0.0f;
        b1f[h] = (M < OUT) ? (_Float16)w1 : (_Float16)0.0f;
    }

    v8f c = {0.f, 0.f, 0.f, 0.f, 0.f, 0.f, 0.f, 0.f};
    c = __builtin_amdgcn_wmma_f32_16x16x32_f16(false, a0, false, b0f,
                                               (short)0, c, false, false);
    c = __builtin_amdgcn_wmma_f32_16x16x32_f16(false, a1, false, b1f,
                                               (short)0, c, false, false);

    const float bias = (M < OUT) ? fc_b[M] : 0.0f;
    #pragma unroll
    for (int r = 0; r < 8; ++r) {
        int Mr = r + (hi ? 8 : 0);
        if (M < OUT)
            out[(size_t)(m0 + Mr) * OUT + M] = c[r] + bias;
    }
}

// ---------------------------------------------------------------------------
// Launch
// ---------------------------------------------------------------------------
extern "C" void kernel_launch(void* const* d_in, const int* in_sizes, int n_in,
                              void* d_out, int out_size, void* d_ws, size_t ws_size,
                              hipStream_t stream) {
    const float* x        = (const float*)d_in[0];
    const float* input_w  = (const float*)d_in[1];
    const float* input_b  = (const float*)d_in[2];
    const float* sw       = (const float*)d_in[3];
    const float* smu      = (const float*)d_in[4];
    const float* ssg      = (const float*)d_in[5];
    const float* serev    = (const float*)d_in[6];
    const float* w        = (const float*)d_in[7];
    const float* mu       = (const float*)d_in[8];
    const float* sg       = (const float*)d_in[9];
    const float* erev     = (const float*)d_in[10];
    const float* gleak    = (const float*)d_in[11];
    const float* vleak    = (const float*)d_in[12];
    const float* cm       = (const float*)d_in[13];
    const float* out_w    = (const float*)d_in[14];
    const float* out_b    = (const float*)d_in[15];
    const float* ln_w     = (const float*)d_in[16];
    const float* ln_b     = (const float*)d_in[17];
    const float* fc_w     = (const float*)d_in[18];
    const float* fc_b     = (const float*)d_in[19];

    // workspace layout (16B aligned)
    char* ws = (char*)d_ws;
    float4*   sens4 = (float4*)(ws);                          // 128 KB
    float4*   rec4  = (float4*)(ws + 131072);                 //  64 KB
    float4*   neur4 = (float4*)(ws + 131072 + 65536);         //   1 KB
    _Float16* hln   = (_Float16*)(ws + 131072 + 65536 + 1024);//  64 KB

    {
        int total = IN * UNITS + UNITS * UNITS + UNITS;       // 12352
        ltc_prep_kernel<<<(total + 255) / 256, 256, 0, stream>>>(
            input_w, input_b, sw, smu, ssg, serev,
            w, mu, sg, erev, gleak, vleak, cm,
            sens4, rec4, neur4);
    }

    ltc_scan_kernel<<<BATCH / BT, 1024, 0, stream>>>(
        x, sens4, rec4, neur4, out_w, out_b, ln_w, ln_b, hln);

    ltc_head_wmma_kernel<<<BATCH / 16, 32, 0, stream>>>(hln, fc_w, fc_b,
                                                        (float*)d_out);
}
